// Attention_22840636080830
// MI455X (gfx1250) — matrix operand order repaired and checked by reading, not compiled, not run on hardware
//
#include <hip/hip_runtime.h>
#include <hip/hip_bf16.h>

typedef __bf16 v16bf __attribute__((ext_vector_type(16)));
typedef __bf16 v8bf  __attribute__((ext_vector_type(8)));
typedef float  v8f   __attribute__((ext_vector_type(8)));

#define D_EMBED 1024
#define D_HEAD  128
#define SEQ     4096
#define NBATCH  4

#define QKV_ELEMS ((size_t)NBATCH * SEQ * D_HEAD)      // per matrix, 2,097,152
#define W_ELEMS   ((size_t)D_HEAD * D_EMBED)           // per matrix, 131,072

#define SCALE 0.0027621358f                            // 1/(sqrt(128)*sqrt(1024))

__device__ __forceinline__ v8f zero8() {
    v8f z;
#pragma unroll
    for (int i = 0; i < 8; ++i) z[i] = 0.0f;
    return z;
}

// Build a 16x32 bf16 A-fragment from its two contiguous 8-element runs
// (elements 0-7 <-> K = 16h+0..7, elements 8-15 <-> K = 16h+8..15).
__device__ __forceinline__ v16bf load_a_frag(const __bf16* p0, const __bf16* p1) {
    union { v16bf v; v8bf h[2]; } u;
    u.h[0] = *(const v8bf*)p0;
    u.h[1] = *(const v8bf*)p1;
    return u.v;
}

// ---------------------------------------------------------------------------
// Kernel 0: convert Wq|Wk|Wv (f32) -> contiguous bf16 once (768 KB).
// ---------------------------------------------------------------------------
__global__ __launch_bounds__(256)
void wconv_kernel(const float* __restrict__ Wq,
                  const float* __restrict__ Wk,
                  const float* __restrict__ Wv,
                  __bf16* __restrict__ wbf)
{
    const size_t i8 = ((size_t)blockIdx.x * 256 + threadIdx.x) * 8;
    const int    m  = (int)(i8 / W_ELEMS);
    const size_t off = i8 % W_ELEMS;
    const float* src = ((m == 0) ? Wq : (m == 1) ? Wk : Wv) + off;
    v8bf d;
#pragma unroll
    for (int e = 0; e < 8; ++e) d[e] = (__bf16)src[e];
    *(v8bf*)(wbf + i8) = d;
}

// ---------------------------------------------------------------------------
// Kernel 1: fused QKV projection.  One wave = 16 rows -> q,k,v (x read ONCE).
// q stored pre-scaled by SCALE; q,k stored [B][S][D]; v stored TRANSPOSED
// [B][D][S] so kernel 2's PV B-fragments are contiguous loads.
// 4*256 = 1024 wave tasks.
// ---------------------------------------------------------------------------
__global__ __launch_bounds__(256)
void qkv_proj_kernel(const float* __restrict__ x,
                     const __bf16* __restrict__ wbf,
                     __bf16* __restrict__ qkv)
{
    const int lane = threadIdx.x & 31;
    const int widx = threadIdx.x >> 5;
    const int task = blockIdx.x * 8 + widx;      // 0..1023
    const int b    = task >> 8;                  // batch
    const int rt   = task & 255;                 // 16-row tile within batch
    const int h    = lane >> 4;
    const int l16  = lane & 15;

    const int   arow = rt * 16 + l16;            // this lane's A row
    const float* xrow = x + ((size_t)b * SEQ + arow) * D_EMBED;

    v8f acc[3][8];
#pragma unroll
    for (int m = 0; m < 3; ++m)
#pragma unroll
        for (int t = 0; t < 8; ++t) acc[m][t] = zero8();

    for (int kb = 0; kb < D_EMBED; kb += 32) {
        // A fragment: 16x32 bf16, rows of x (f32 -> bf16 on the fly)
        v16bf a;
#pragma unroll
        for (int e = 0; e < 8; ++e) a[e]     = (__bf16)xrow[kb + h * 16 + e];
#pragma unroll
        for (int e = 0; e < 8; ++e) a[8 + e] = (__bf16)xrow[kb + h * 16 + 8 + e];

        // 24 pipelined (mat, nt) B-fragment loads + WMMAs
        const __bf16* wchunk = wbf + (size_t)l16 * D_EMBED + kb + h * 16;
        v16bf bcur = *(const v16bf*)(wchunk);
#pragma unroll
        for (int i = 0; i < 24; ++i) {
            if (i < 23) {
                const int j   = i + 1;
                const int mat = j >> 3, nt = j & 7;
                const v16bf bnext = *(const v16bf*)(wchunk
                    + (size_t)mat * W_ELEMS + (size_t)nt * 16 * D_EMBED);
                acc[i >> 3][i & 7] = __builtin_amdgcn_wmma_f32_16x16x32_bf16(
                    false, a, false, bcur, (short)0, acc[i >> 3][i & 7], false, false);
                bcur = bnext;
            } else {
                acc[i >> 3][i & 7] = __builtin_amdgcn_wmma_f32_16x16x32_bf16(
                    false, a, false, bcur, (short)0, acc[i >> 3][i & 7], false, false);
            }
        }
    }

    // q (pre-scaled) and k: [B][S][D]
    __bf16* qo = qkv;
    __bf16* ko = qkv + QKV_ELEMS;
    __bf16* vt = qkv + 2 * QKV_ELEMS;            // [B][D][S]
#pragma unroll
    for (int nt = 0; nt < 8; ++nt)
#pragma unroll
        for (int r = 0; r < 8; ++r) {
            const int orow = rt * 16 + r + 8 * h;
            const int col  = nt * 16 + l16;
            const size_t rowmaj = ((size_t)b * SEQ + orow) * D_HEAD + col;
            qo[rowmaj] = (__bf16)(acc[0][nt][r] * SCALE);
            ko[rowmaj] = (__bf16)acc[1][nt][r];
            vt[((size_t)b * D_HEAD + col) * SEQ + orow] = (__bf16)acc[2][nt][r];
        }
}

// ---------------------------------------------------------------------------
// Kernel 2: flash attention.  Each wave owns 16 query rows (full d_head=128),
// streams keys/values in blocks of 64, online softmax in f32.
// Scale is pre-folded into q.
// ---------------------------------------------------------------------------
__global__ __launch_bounds__(256)
void flash_attn_kernel(const __bf16* __restrict__ qkv,
                       float* __restrict__ out)
{
    const int lane = threadIdx.x & 31;
    const int widx = threadIdx.x >> 5;
    const int h    = lane >> 4;
    const int l16  = lane & 15;

    const int wg_per_batch = SEQ / 128;          // 32
    const int b     = blockIdx.x / wg_per_batch;
    const int qblk  = blockIdx.x % wg_per_batch;
    const int qbase = qblk * 128 + widx * 16;

    const __bf16* qb  = qkv;
    const __bf16* kbm = qkv + QKV_ELEMS;
    const __bf16* vtb = qkv + 2 * QKV_ELEMS + (size_t)b * D_HEAD * SEQ; // [D][S]

    // Q fragments: 4 chunks of K=32 over d_head
    const __bf16* qrow = qb + ((size_t)b * SEQ + qbase + l16) * D_HEAD;
    v16bf qa[4];
#pragma unroll
    for (int c = 0; c < 4; ++c)
        qa[c] = load_a_frag(qrow + c * 32 + h * 16,
                            qrow + c * 32 + h * 16 + 8);

    v8f o[8];
#pragma unroll
    for (int t = 0; t < 8; ++t) o[t] = zero8();
    float mrow[8], lrow[8];
#pragma unroll
    for (int r = 0; r < 8; ++r) { mrow[r] = -3.0e38f; lrow[r] = 0.0f; }

    __shared__ __bf16 Plds[8][16 * 64];          // per-wave P staging (16 KB)
    __bf16* P = Plds[widx];

    for (int kb0 = 0; kb0 < SEQ; kb0 += 64) {
        const __bf16* kbase = kbm + ((size_t)b * SEQ + kb0) * D_HEAD;

        if (kb0 + 64 < SEQ) {                    // best-effort prefetch
            __builtin_prefetch(kbase + (size_t)(64 + lane) * D_HEAD, 0, 0);
            __builtin_prefetch(kbase + (size_t)(96 + lane) * D_HEAD, 0, 0);
#pragma unroll
            for (int pp = 0; pp < 4; ++pp)
                __builtin_prefetch(vtb + (size_t)(lane + 32 * pp) * SEQ + kb0 + 64, 0, 0);
        }

        // S = Q K^T for 64 keys: four 16x16 D tiles, loads pipelined 1 chunk ahead
        v8f s[4];
#pragma unroll
        for (int st = 0; st < 4; ++st) s[st] = zero8();

        v16bf bk[4];
#pragma unroll
        for (int st = 0; st < 4; ++st)
            bk[st] = *(const v16bf*)(kbase + (size_t)(st * 16 + l16) * D_HEAD + h * 16);
#pragma unroll
        for (int c = 0; c < 4; ++c) {
            v16bf bkn[4];
            if (c < 3) {
#pragma unroll
                for (int st = 0; st < 4; ++st)
                    bkn[st] = *(const v16bf*)(kbase
                        + (size_t)(st * 16 + l16) * D_HEAD + (c + 1) * 32 + h * 16);
            }
#pragma unroll
            for (int st = 0; st < 4; ++st)
                s[st] = __builtin_amdgcn_wmma_f32_16x16x32_bf16(
                    false, qa[c], false, bk[st], (short)0, s[st], false, false);
            if (c < 3) {
#pragma unroll
                for (int st = 0; st < 4; ++st) bk[st] = bkn[st];
            }
        }

        // online softmax (rows live across the 16 lanes of each half)
        float mx[8];
#pragma unroll
        for (int r = 0; r < 8; ++r) {
            float m01 = fmaxf(s[0][r], s[1][r]);
            float m23 = fmaxf(s[2][r], s[3][r]);
            mx[r] = fmaxf(m01, m23);
        }
#pragma unroll
        for (int off = 1; off <= 8; off <<= 1)
#pragma unroll
            for (int r = 0; r < 8; ++r)
                mx[r] = fmaxf(mx[r], __shfl_xor(mx[r], off, 32));

        float alpha[8], rs[8];
#pragma unroll
        for (int r = 0; r < 8; ++r) {
            const float mn = fmaxf(mrow[r], mx[r]);
            alpha[r] = __expf(mrow[r] - mn);
            mrow[r]  = mn;
            rs[r] = 0.0f;
#pragma unroll
            for (int st = 0; st < 4; ++st) {
                s[st][r] = __expf(s[st][r] - mn);
                rs[r] += s[st][r];
            }
        }
#pragma unroll
        for (int off = 1; off <= 8; off <<= 1)
#pragma unroll
            for (int r = 0; r < 8; ++r)
                rs[r] += __shfl_xor(rs[r], off, 32);
#pragma unroll
        for (int r = 0; r < 8; ++r) lrow[r] = lrow[r] * alpha[r] + rs[r];
#pragma unroll
        for (int t = 0; t < 8; ++t)
#pragma unroll
            for (int r = 0; r < 8; ++r) o[t][r] *= alpha[r];

        // P: D-layout -> A-layout through per-wave LDS (row stride 64 keys)
#pragma unroll
        for (int st = 0; st < 4; ++st)
#pragma unroll
            for (int r = 0; r < 8; ++r)
                P[(r + 8 * h) * 64 + st * 16 + l16] = (__bf16)s[st][r];
        asm volatile("s_wait_dscnt 0" ::: "memory");

        v16bf pa[2];
#pragma unroll
        for (int f = 0; f < 2; ++f)
            pa[f] = load_a_frag(P + l16 * 64 + 32 * f + h * 16,
                                P + l16 * 64 + 32 * f + h * 16 + 8);

        // O += P V : pipelined contiguous loads from V^T
        v16bf bv[2];
#pragma unroll
        for (int f = 0; f < 2; ++f)
            bv[f] = *(const v16bf*)(vtb + (size_t)l16 * SEQ + kb0 + 32 * f + 16 * h);
#pragma unroll
        for (int t = 0; t < 8; ++t) {
            v16bf bvn[2];
            if (t < 7) {
#pragma unroll
                for (int f = 0; f < 2; ++f)
                    bvn[f] = *(const v16bf*)(vtb
                        + (size_t)((t + 1) * 16 + l16) * SEQ + kb0 + 32 * f + 16 * h);
            }
#pragma unroll
            for (int f = 0; f < 2; ++f)
                o[t] = __builtin_amdgcn_wmma_f32_16x16x32_bf16(
                    false, pa[f], false, bv[f], (short)0, o[t], false, false);
            if (t < 7) {
#pragma unroll
                for (int f = 0; f < 2; ++f) bv[f] = bvn[f];
            }
        }
    }

    // normalize + store (f32 output)
    float inv[8];
#pragma unroll
    for (int r = 0; r < 8; ++r) inv[r] = 1.0f / lrow[r];
#pragma unroll
    for (int t = 0; t < 8; ++t)
#pragma unroll
        for (int r = 0; r < 8; ++r) {
            const int orow = qbase + r + 8 * h;
            out[((size_t)b * SEQ + orow) * D_HEAD + t * 16 + l16] = o[t][r] * inv[r];
        }
}

extern "C" void kernel_launch(void* const* d_in, const int* in_sizes, int n_in,
                              void* d_out, int out_size, void* d_ws, size_t ws_size,
                              hipStream_t stream) {
    (void)in_sizes; (void)n_in; (void)out_size; (void)ws_size;
    const float* x  = (const float*)d_in[0];
    const float* Wq = (const float*)d_in[1];
    const float* Wk = (const float*)d_in[2];
    const float* Wv = (const float*)d_in[3];

    __bf16* qkv = (__bf16*)d_ws;                     // 3 * 2M bf16 = 12 MB
    __bf16* wbf = qkv + 3 * QKV_ELEMS;               // 3 * 128K bf16 = 768 KB

    // 3*131072 elems / (256 thr * 8) = 192 blocks
    wconv_kernel<<<192, 256, 0, stream>>>(Wq, Wk, Wv, wbf);
    // 1024 fused wave tasks / 8 waves per block
    qkv_proj_kernel<<<128, 256, 0, stream>>>(x, wbf, qkv);
    // 4 batches * 32 query blocks of 128 rows
    flash_attn_kernel<<<128, 256, 0, stream>>>(qkv, (float*)d_out);
}
